// DensityDecoderRNN_88776974008559
// MI455X (gfx1250) — matrix operand order repaired and checked by reading, not compiled, not run on hardware
//
#include <hip/hip_runtime.h>
#include <cmath>

// ---------------------------------------------------------------------------
// DensityDecoderRNN on MI455X (gfx1250): bf16 WMMA + TDM double-buffering,
// all weights pre-transposed so every WMMA fragment is a contiguous b128 run.
//   B=32, T=512, D=1024, H=1024, L=2, F=256
//  - x_proj GEMM: 128x128 tiles, ping-pong LDS, tensor_load_to_lds prefetch
//    of stage k+1 overlapped with WMMA on stage k (s_wait_tensorcnt 2 / 0)
//  - recurrence: ONE persistent kernel per layer (32 WGs), global software
//    barrier (agent-scope atomics + s_sleep), c in registers, h & Wh^T tiles
//    TDM double-buffered (Wh^T via 3D gate-strided descriptor)
//  - fused dual-head GEMM (softplus epilogue) straight into d_out
// ---------------------------------------------------------------------------

typedef __bf16 bf16;
typedef float        v8f   __attribute__((ext_vector_type(8)));
typedef bf16         v16bf __attribute__((ext_vector_type(16)));
typedef unsigned int u32x4 __attribute__((ext_vector_type(4)));
typedef int          i32x4 __attribute__((ext_vector_type(4)));
typedef int          i32x8 __attribute__((ext_vector_type(8)));

#define WMMA_BF16(a, b, c) \
  __builtin_amdgcn_wmma_f32_16x16x32_bf16(false, (a), false, (b), (short)0, (c), false, false)

#if defined(__HIP_DEVICE_COMPILE__) && __has_builtin(__builtin_amdgcn_tensor_load_to_lds)
#define HAVE_TDM 1
#else
#define HAVE_TDM 0
#endif

constexpr int Bn = 32, Tn = 512, Dn = 1024, Hn = 1024, Fn = 256, G4 = 4096;
constexpr int Mrows = Bn * Tn;  // 16384
constexpr int LSTM_WGS = 32;    // persistent grid: one WG per 32 hidden units

__device__ __forceinline__ float sigmoidf_(float x) { return 1.0f / (1.0f + __expf(-x)); }
__device__ __forceinline__ float softplusf_(float x) { return (x > 20.0f) ? x : log1pf(__expf(x)); }

// --------------------------- elementwise helpers ---------------------------

__global__ void cvt_f32_to_bf16(const float* __restrict__ in, bf16* __restrict__ out, long n) {
  long i = (long)blockIdx.x * blockDim.x + threadIdx.x;
  long stride = (long)gridDim.x * blockDim.x;
  for (; i < n; i += stride) out[i] = (bf16)in[i];
}

// Tiled transpose + convert: in [R][C] f32 -> out [C][R] bf16. R,C % 32 == 0.
__global__ __launch_bounds__(256)
void cvt_transpose_f32_bf16(const float* __restrict__ in, bf16* __restrict__ out,
                            int R, int C) {
  __shared__ float tile[32][33];
  const int c0 = blockIdx.x * 32, r0 = blockIdx.y * 32;
  const int tx = threadIdx.x & 31, ty = threadIdx.x >> 5;  // 32 x 8
#pragma unroll
  for (int i = 0; i < 32; i += 8)
    tile[ty + i][tx] = in[(size_t)(r0 + ty + i) * C + (c0 + tx)];
  __syncthreads();
#pragma unroll
  for (int i = 0; i < 32; i += 8)
    out[(size_t)(c0 + ty + i) * R + (r0 + tx)] = (bf16)tile[tx][ty + i];
}

__global__ void zero_bf16_k(bf16* __restrict__ p, int n) {
  int i = blockIdx.x * blockDim.x + threadIdx.x;
  if (i < n) p[i] = (bf16)0.0f;
}

__global__ void zero_u32_k(unsigned* __restrict__ p, int n) {
  int i = blockIdx.x * blockDim.x + threadIdx.x;
  if (i < n) p[i] = 0u;
}

// ------------------- WMMA fragment loaders (from LDS) ----------------------
// A (16x32 bf16): lane l holds row M=l&15; K = 16*(l>>4) + e
// B (32x16 bf16): lane l holds col N=l&15; K = 16*(l>>4) + e
// C/D (16x16 f32): lane l col N=l&15; vgpr r row M = r + 8*(l>>4)
// Both A and B rows live row-major-by-(M or N) in LDS with 16B-aligned
// pitch 40, so every fragment is 1-2 contiguous 16B runs -> ds_load_b128.

__device__ __forceinline__ v16bf load_a_frag(const bf16* __restrict__ row, int lh) {
  v16bf a;
#pragma unroll
  for (int e = 0; e < 16; ++e) a[e] = row[lh * 16 + e];
  return a;
}

__device__ __forceinline__ v16bf load_b_frag(const bf16* __restrict__ row, int lh) {
  v16bf b;
#pragma unroll
  for (int e = 0; e < 16; ++e) b[e] = row[lh * 16 + e];
  return b;
}

// --------------------------- TDM descriptors -------------------------------
// pad codes: interval v -> 2^(v+1) DWORDs of data between pads,
//            amount  a -> a+1 DWORDs of pad inserted.
// All tiles here: x = 32 contiguous bf16 (16 DW line, pad code 3) with
// pad_amount code 3 (4 DW) -> LDS pitch 40 bf16 (80 B, 16B aligned).

#if HAVE_TDM
// 2D tile: x = tile_x contiguous elems, y = tile_y rows @ stride_e.
__device__ __forceinline__ void tdm_load_2d(const bf16* gptr, unsigned lds_off,
                                            unsigned tile_x, unsigned tile_y,
                                            unsigned stride_e, unsigned dim_x, unsigned dim_y,
                                            unsigned pad_int, unsigned pad_amt) {
  unsigned long long ga = (unsigned long long)(size_t)gptr;
  u32x4 g0;
  g0[0] = 1u;                                                  // count=1, user mode
  g0[1] = lds_off;                                             // lds_addr
  g0[2] = (unsigned)(ga & 0xFFFFFFFFull);                      // global_addr lo
  g0[3] = (unsigned)((ga >> 32) & 0x1FFFFFFull) | (2u << 30);  // addr[56:32] | type=2
  i32x8 g1;
  g1[0] = (int)((1u << 16) | (1u << 20) | (pad_int << 22) | (pad_amt << 25));
  g1[1] = (int)((dim_x & 0xFFFFu) << 16);                      // tensor_dim0 lo16
  g1[2] = (int)((dim_x >> 16) | ((dim_y & 0xFFFFu) << 16));    // dim0 hi | dim1 lo
  g1[3] = (int)((dim_y >> 16) | (tile_x << 16));               // dim1 hi | tile_dim0
  g1[4] = (int)tile_y;                                         // tile_dim1 (tile_dim2=0)
  g1[5] = (int)stride_e;                                       // tensor_dim0_stride lo32
  g1[6] = 0;
  g1[7] = 0;
  i32x4 gz = {0, 0, 0, 0};
#if __clang_major__ >= 23
  i32x8 gx = {};
  __builtin_amdgcn_tensor_load_to_lds(g0, g1, gz, gz, gx, 0);
#else
  __builtin_amdgcn_tensor_load_to_lds(g0, g1, gz, gz, 0);
#endif
}

// Wh^T tile, 3D: x = 32 K contig, y = 32 unit-rows @ stride H,
// z = 4 gates @ stride H*H. LDS: [gate][n][40] (pad 4DW / 16DW).
__device__ __forceinline__ void tdm_load_wht(const bf16* gptr, unsigned lds_off) {
  unsigned long long ga = (unsigned long long)(size_t)gptr;
  const unsigned gstride = (unsigned)Hn * (unsigned)Hn;        // 1<<20 elements
  u32x4 g0;
  g0[0] = 1u;
  g0[1] = lds_off;
  g0[2] = (unsigned)(ga & 0xFFFFFFFFull);
  g0[3] = (unsigned)((ga >> 32) & 0x1FFFFFFull) | (2u << 30);
  i32x8 g1;
  g1[0] = (int)((1u << 16) | (1u << 20) | (3u << 22) | (3u << 25));
  g1[1] = (int)(((unsigned)Hn & 0xFFFFu) << 16);               // tensor_dim0 = H lo16
  g1[2] = (int)(((unsigned)Hn >> 16) | (((unsigned)Hn & 0xFFFFu) << 16));  // | dim1 = H
  g1[3] = (int)(((unsigned)Hn >> 16) | (32u << 16));           // dim1 hi | tile_dim0 = 32
  g1[4] = (int)(32u | (4u << 16));                             // tile_dim1=32 | tile_dim2=4
  g1[5] = (int)(unsigned)Hn;                                   // dim0_stride = H
  g1[6] = (int)((gstride & 0xFFFFu) << 16);                    // dim1_stride lo16 (H*H)
  g1[7] = (int)(gstride >> 16);                                // dim1_stride [47:16]
  i32x4 g2;
  g2[0] = 4; g2[1] = 0; g2[2] = 0; g2[3] = 0;                  // tensor_dim2 = 4 gates
  i32x4 g3 = {0, 0, 0, 0};
#if __clang_major__ >= 23
  i32x8 gx = {};
  __builtin_amdgcn_tensor_load_to_lds(g0, g1, g2, g3, gx, 0);
#else
  __builtin_amdgcn_tensor_load_to_lds(g0, g1, g2, g3, 0);
#endif
}
#endif  // HAVE_TDM

// ------------------- x-projection GEMM: C = A*W^T' + bias (f32 out) --------
// A [M][K] row-major, Wt [N][K] row-major (pre-transposed weights).
// 128x128 block tile, 8 waves x (2x4) 16x16 tiles, TDM double-buffered K.

struct GemmSmem {
  bf16 As[2][128][40];   // [buf][m][k]  pitch 40  20480 B
  bf16 Bs[2][128][40];   // [buf][n][k]  pitch 40  20480 B
};

__global__ __launch_bounds__(256)
void gemm_bf16_bias_f32(const bf16* __restrict__ A, const bf16* __restrict__ Wt,
                        const float* __restrict__ bias, float* __restrict__ C,
                        int M, int N, int K) {
  __shared__ GemmSmem sm;

  const int tid  = threadIdx.x;
  const int wave = tid >> 5, lane = tid & 31;
  const int lm = lane & 15, lh = lane >> 4;
  const int m0 = blockIdx.x * 128, n0 = blockIdx.y * 128;
  const int wm = wave >> 1;        // 0..3 -> Mtiles {2wm, 2wm+1}
  const int wn = wave & 1;         // 0..1 -> Ntiles {4wn .. 4wn+3}
  const int nIter = K / 32;

  v8f acc[2][4] = {};

#if HAVE_TDM
  unsigned aOff[2] = {(unsigned)(size_t)&sm.As[0][0][0], (unsigned)(size_t)&sm.As[1][0][0]};
  unsigned bOff[2] = {(unsigned)(size_t)&sm.Bs[0][0][0], (unsigned)(size_t)&sm.Bs[1][0][0]};
  if (wave == 0) {  // preload stage 0
    tdm_load_2d(&A[(size_t)m0 * K],  aOff[0], 32, 128, K, K, M, 3, 3);
    tdm_load_2d(&Wt[(size_t)n0 * K], bOff[0], 32, 128, K, K, N, 3, 3);
  }
#endif

  for (int it = 0; it < nIter; ++it) {
    const int buf = it & 1;
    const int k0  = it * 32;
#if HAVE_TDM
    if (wave == 0) {
      if (it + 1 < nIter) {   // prefetch next stage into other buffer
        const int k1 = k0 + 32;
        tdm_load_2d(&A[(size_t)m0 * K + k1],  aOff[buf ^ 1], 32, 128, K, K, M, 3, 3);
        tdm_load_2d(&Wt[(size_t)n0 * K + k1], bOff[buf ^ 1], 32, 128, K, K, N, 3, 3);
        __builtin_amdgcn_s_wait_tensorcnt(2);   // current stage's 2 loads done
      } else {
        __builtin_amdgcn_s_wait_tensorcnt(0);
      }
    }
#else
    {  // fallback: thread staging, 16-elem contiguous runs
      const int sr = tid >> 1, sc = (tid & 1) * 16;
      const bf16* ga = &A[(size_t)(m0 + sr) * K + k0 + sc];
      const bf16* gb = &Wt[(size_t)(n0 + sr) * K + k0 + sc];
      bf16* la = &sm.As[buf][sr][sc];
      bf16* lb = &sm.Bs[buf][sr][sc];
#pragma unroll
      for (int e = 0; e < 16; ++e) { la[e] = ga[e]; lb[e] = gb[e]; }
    }
#endif
    __syncthreads();

    v16bf a0 = load_a_frag(&sm.As[buf][(wm * 2 + 0) * 16 + lm][0], lh);
    v16bf a1 = load_a_frag(&sm.As[buf][(wm * 2 + 1) * 16 + lm][0], lh);
#pragma unroll
    for (int j = 0; j < 4; ++j) {
      v16bf bb = load_b_frag(&sm.Bs[buf][(wn * 4 + j) * 16 + lm][0], lh);
      acc[0][j] = WMMA_BF16(a0, bb, acc[0][j]);
      acc[1][j] = WMMA_BF16(a1, bb, acc[1][j]);
    }
    __syncthreads();
  }

#pragma unroll
  for (int i = 0; i < 2; ++i)
#pragma unroll
    for (int j = 0; j < 4; ++j)
#pragma unroll
      for (int r = 0; r < 8; ++r) {
        int m = m0 + (wm * 2 + i) * 16 + r + lh * 8;
        int n = n0 + (wn * 4 + j) * 16 + lm;
        C[(size_t)m * N + n] = acc[i][j][r] + bias[n];
      }
}

// ----------------------- persistent LSTM layer kernel ----------------------
// Grid: exactly LSTM_WGS (=32) workgroups x 256 threads; WG g owns hidden
// units [32g, 32g+32). Must be co-resident (32 WGs << WGP count).

struct LstmSmem {
  bf16  Ah[2][32][40];     // h tile [buf][b][k]                       5120 B
  bf16  Bw[2][4][32][40];  // Wh^T tile [buf][gate][n][k]             20480 B
  float Gs[4][32][33];     // gate preactivations                     16896 B
};

__global__ __launch_bounds__(256)
void lstm_layer_persistent(const float* __restrict__ xproj,   // [B*T, 4H]
                           const bf16*  __restrict__ WhT,     // [4H, H] transposed
                           bf16*        __restrict__ hs,      // [B, T, H] out
                           bf16*        __restrict__ h_a,     // [B, H] ping (pre-zeroed)
                           bf16*        __restrict__ h_b,     // [B, H] pong
                           unsigned*    __restrict__ barrier_ctr) {
  __shared__ LstmSmem sm;

  const int tid  = threadIdx.x;
  const int wave = tid >> 5, lane = tid & 31;
  const int lm = lane & 15, lh = lane >> 4;
  const int n0 = blockIdx.x * 32;
  const int gate  = wave >> 1;   // 0..3 = i,f,g,o
  const int mtile = wave & 1;    // batch rows 0-15 / 16-31
  const int nK = Hn / 32;

  // this thread's 4 owned (b, n) cells: e = tid + 256j -> b = e>>5, nl = e&31
  float creg[4] = {0.f, 0.f, 0.f, 0.f};

#if HAVE_TDM
  unsigned ahOff[2] = {(unsigned)(size_t)&sm.Ah[0][0][0], (unsigned)(size_t)&sm.Ah[1][0][0]};
  unsigned bwOff[2] = {(unsigned)(size_t)&sm.Bw[0][0][0][0], (unsigned)(size_t)&sm.Bw[1][0][0][0]};
#endif

  for (int t = 0; t < Tn; ++t) {
    const bf16* hin  = (t & 1) ? h_b : h_a;
    bf16*       hout = (t & 1) ? h_a : h_b;

    // prefetch next timestep's x_proj slice for this WG
    if (t + 1 < Tn) {
      int m = tid >> 3, s = tid & 7, g = s >> 1, off = (s & 1) * 16;
      __builtin_prefetch(&xproj[((size_t)m * Tn + (t + 1)) * G4 + g * Hn + n0 + off], 0, 0);
    }

    v8f acc0 = {}, acc1 = {};

#if HAVE_TDM
    if (wave == 0) {  // preload stage 0 (h rows + Wh^T tile)
      tdm_load_2d(hin, ahOff[0], 32, 32, Hn, Hn, Bn, 3, 3);
      tdm_load_wht(WhT + (size_t)n0 * Hn, bwOff[0]);
    }
#endif

    for (int ki = 0; ki < nK; ++ki) {
      const int buf = ki & 1;
      const int k0  = ki * 32;
#if HAVE_TDM
      if (wave == 0) {
        if (ki + 1 < nK) {
          const int k1 = k0 + 32;
          tdm_load_2d(hin + k1, ahOff[buf ^ 1], 32, 32, Hn, Hn, Bn, 3, 3);
          tdm_load_wht(WhT + (size_t)n0 * Hn + k1, bwOff[buf ^ 1]);
          __builtin_amdgcn_s_wait_tensorcnt(2);
        } else {
          __builtin_amdgcn_s_wait_tensorcnt(0);
        }
      }
#else
      {
#pragma unroll
        for (int j = 0; j < 4; ++j) {               // stage h: 32x32
          int e = tid + j * 256;
          int r = e >> 5, cc = e & 31;
          sm.Ah[buf][r][cc] = hin[r * Hn + (k0 + cc)];
        }
        {  // stage Wh^T: 128 rows x 32 k, contiguous 16-elem runs
          const int sr = tid >> 1, sc = (tid & 1) * 16;
          const int g = sr >> 5, nn = sr & 31;
          const bf16* gsrc = &WhT[(size_t)(g * Hn + n0 + nn) * Hn + k0 + sc];
          bf16* l = &sm.Bw[buf][g][nn][sc];
#pragma unroll
          for (int e = 0; e < 16; ++e) l[e] = gsrc[e];
        }
      }
#endif
      __syncthreads();

      v16bf af = load_a_frag(&sm.Ah[buf][mtile * 16 + lm][0], lh);
      v16bf b0 = load_b_frag(&sm.Bw[buf][gate][lm][0], lh);
      v16bf b1 = load_b_frag(&sm.Bw[buf][gate][16 + lm][0], lh);
      acc0 = WMMA_BF16(af, b0, acc0);
      acc1 = WMMA_BF16(af, b1, acc1);
      __syncthreads();
    }

    // add x_proj, park gate preactivations for cross-wave combine
#pragma unroll
    for (int r = 0; r < 8; ++r) {
      int m = mtile * 16 + r + lh * 8;             // batch index
      size_t row = ((size_t)m * Tn + t) * G4;
      int col0 = gate * Hn + n0 + lm;
      sm.Gs[gate][m][lm]      = acc0[r] + xproj[row + col0];
      sm.Gs[gate][m][lm + 16] = acc1[r] + xproj[row + col0 + 16];
    }
    __syncthreads();

#pragma unroll
    for (int j = 0; j < 4; ++j) {
      int e = tid + j * 256;
      int m = e >> 5, nl = e & 31;
      int n = n0 + nl;
      float ig = sigmoidf_(sm.Gs[0][m][nl]);
      float fg = sigmoidf_(sm.Gs[1][m][nl]);
      float gg = tanhf(sm.Gs[2][m][nl]);
      float og = sigmoidf_(sm.Gs[3][m][nl]);
      float cn = fg * creg[j] + ig * gg;
      creg[j] = cn;
      bf16 hb = (bf16)(og * tanhf(cn));
      hout[m * Hn + n] = hb;
      hs[((size_t)m * Tn + t) * Hn + n] = hb;
    }

    // grid-wide software barrier (release h writes, acquire others')
    __threadfence();
    __syncthreads();
    if (tid == 0) {
      __hip_atomic_fetch_add(barrier_ctr, 1u, __ATOMIC_ACQ_REL, __HIP_MEMORY_SCOPE_AGENT);
      unsigned target = (unsigned)LSTM_WGS * (unsigned)(t + 1);
      while (__hip_atomic_load(barrier_ctr, __ATOMIC_ACQUIRE, __HIP_MEMORY_SCOPE_AGENT) < target)
        __builtin_amdgcn_s_sleep(1);
    }
    __syncthreads();
  }
}

// ------------------------ fused output heads -------------------------------
// blockIdx.y in [0,8): 0..3 -> mu head (softplus + y_min), 4..7 -> log_sigma.
// Weights pre-transposed: WT [F][H] row-major.

__global__ __launch_bounds__(256)
void head_gemm(const bf16* __restrict__ A,
               const bf16* __restrict__ WmuT, const bf16* __restrict__ WlsT,
               const float* __restrict__ bmu, const float* __restrict__ bls,
               const float* __restrict__ ymin,
               float* __restrict__ mu, float* __restrict__ ls) {
  __shared__ bf16 As[64][40];
  __shared__ bf16 Bt[64][40];

  const int tid  = threadIdx.x;
  const int wave = tid >> 5, lane = tid & 31;
  const int lm = lane & 15, lh = lane >> 4;
  const int m0 = blockIdx.x * 64;
  const bool is_mu = (blockIdx.y < 4);
  const int  n0 = (blockIdx.y & 3) * 64;
  const bf16*  WT   = is_mu ? WmuT : WlsT;
  const float* bias = is_mu ? bmu : bls;
  float* out = is_mu ? mu : ls;

  const int wm = wave >> 1;
  const int wn = (wave & 1) * 2;

  v8f acc0 = {}, acc1 = {};

  const int sr = tid >> 2, sc = (tid & 3) * 8;   // 64 rows x 32 k, 8-elem runs

  for (int k0 = 0; k0 < Hn; k0 += 32) {
    {
      const bf16* ga = &A[(size_t)(m0 + sr) * Hn + k0 + sc];
      const bf16* gb = &WT[(size_t)(n0 + sr) * Hn + k0 + sc];
      bf16* la = &As[sr][sc];
      bf16* lb = &Bt[sr][sc];
#pragma unroll
      for (int e = 0; e < 8; ++e) { la[e] = ga[e]; lb[e] = gb[e]; }
    }
    __syncthreads();

    v16bf af = load_a_frag(&As[wm * 16 + lm][0], lh);
    v16bf b0 = load_b_frag(&Bt[wn * 16 + lm][0], lh);
    v16bf b1 = load_b_frag(&Bt[(wn + 1) * 16 + lm][0], lh);
    acc0 = WMMA_BF16(af, b0, acc0);
    acc1 = WMMA_BF16(af, b1, acc1);
    __syncthreads();
  }

#pragma unroll
  for (int r = 0; r < 8; ++r) {
    int m = m0 + wm * 16 + r + lh * 8;
    int n = n0 + wn * 16 + lm;
    float v0 = acc0[r] + bias[n];
    float v1 = acc1[r] + bias[n + 16];
    if (is_mu) {
      v0 = softplusf_(v0) + ymin[n];
      v1 = softplusf_(v1) + ymin[n + 16];
    }
    out[(size_t)m * Fn + n]        = v0;
    out[(size_t)m * Fn + (n + 16)] = v1;
  }
}

// ------------------------------- launcher ----------------------------------

extern "C" void kernel_launch(void* const* d_in, const int* in_sizes, int n_in,
                              void* d_out, int out_size, void* d_ws, size_t ws_size,
                              hipStream_t stream) {
  (void)in_sizes; (void)n_in; (void)out_size; (void)ws_size;

  const float* x    = (const float*)d_in[0];
  const float* Wx   = (const float*)d_in[1];
  const float* Wh   = (const float*)d_in[2];
  const float* bg   = (const float*)d_in[3];
  const float* Wmu  = (const float*)d_in[4];
  const float* bmu  = (const float*)d_in[5];
  const float* Wls  = (const float*)d_in[6];
  const float* bls  = (const float*)d_in[7];
  const float* ymin = (const float*)d_in[8];

  float* mu = (float*)d_out;
  float* ls = mu + (size_t)Mrows * Fn;

  char* ws = (char*)d_ws;
  size_t off = 0;
  auto carve = [&](size_t bytes) -> void* {
    void* p = ws + off;
    off += (bytes + 255) & ~(size_t)255;
    return p;
  };
  bf16*     x_bf   = (bf16*)carve((size_t)Mrows * Dn * sizeof(bf16));
  bf16*     hs0_bf = (bf16*)carve((size_t)Mrows * Hn * sizeof(bf16));
  bf16*     WxT    = (bf16*)carve((size_t)2 * G4 * Dn * sizeof(bf16));  // [l][4H][D]
  bf16*     WhT    = (bf16*)carve((size_t)2 * G4 * Hn * sizeof(bf16));  // [l][4H][H]
  bf16*     WmuT   = (bf16*)carve((size_t)Fn * Hn * sizeof(bf16));      // [F][H]
  bf16*     WlsT   = (bf16*)carve((size_t)Fn * Hn * sizeof(bf16));
  float*    xproj  = (float*)carve((size_t)Mrows * G4 * sizeof(float));
  bf16*     h_a    = (bf16*)carve((size_t)Bn * Hn * sizeof(bf16));
  bf16*     h_b    = (bf16*)carve((size_t)Bn * Hn * sizeof(bf16));
  unsigned* ctrs   = (unsigned*)carve(2 * sizeof(unsigned));

  zero_u32_k<<<1, 32, 0, stream>>>(ctrs, 2);

  cvt_f32_to_bf16<<<4096, 256, 0, stream>>>(x, x_bf, (long)Mrows * Dn);
  for (int l = 0; l < 2; ++l) {
    cvt_transpose_f32_bf16<<<dim3(G4 / 32, Dn / 32), 256, 0, stream>>>(
        Wx + (size_t)l * Dn * G4, WxT + (size_t)l * G4 * Dn, Dn, G4);
    cvt_transpose_f32_bf16<<<dim3(G4 / 32, Hn / 32), 256, 0, stream>>>(
        Wh + (size_t)l * Hn * G4, WhT + (size_t)l * G4 * Hn, Hn, G4);
  }
  cvt_transpose_f32_bf16<<<dim3(Fn / 32, Hn / 32), 256, 0, stream>>>(Wmu, WmuT, Hn, Fn);
  cvt_transpose_f32_bf16<<<dim3(Fn / 32, Hn / 32), 256, 0, stream>>>(Wls, WlsT, Hn, Fn);

  for (int l = 0; l < 2; ++l) {
    const bf16* in_bf  = (l == 0) ? x_bf   : hs0_bf;
    bf16*       out_hs = (l == 0) ? hs0_bf : x_bf;   // reuse x_bf for layer-1 hs

    gemm_bf16_bias_f32<<<dim3(Mrows / 128, G4 / 128), 256, 0, stream>>>(
        in_bf, WxT + (size_t)l * G4 * Dn, bg + (size_t)l * G4, xproj,
        Mrows, G4, Dn);

    zero_bf16_k<<<(Bn * Hn + 255) / 256, 256, 0, stream>>>(h_a, Bn * Hn);

    lstm_layer_persistent<<<LSTM_WGS, 256, 0, stream>>>(
        xproj, WhT + (size_t)l * G4 * Hn, out_hs, h_a, h_b, ctrs + l);
  }

  head_gemm<<<dim3(Mrows / 64, 8), 256, 0, stream>>>(
      x_bf, WmuT, WlsT, bmu, bls, ymin, mu, ls);
}
